// RNN_Classifiters_14199161880613
// MI455X (gfx1250) — matrix operand order repaired and checked by reading, not compiled, not run on hardware
//
#include <hip/hip_runtime.h>

typedef __attribute__((ext_vector_type(16))) _Float16 v16h;
typedef __attribute__((ext_vector_type(8)))  _Float16 h8;
typedef __attribute__((ext_vector_type(4)))  _Float16 h4;
typedef __attribute__((ext_vector_type(8)))  float    v8f;
typedef __attribute__((ext_vector_type(4)))  float    f4;

constexpr int kH  = 100;   // hidden size
constexpr int kI  = 64;    // input size per step
constexpr int kT  = 64;    // timesteps
constexpr int kC  = 40;    // classes
constexpr int kHP = 128;   // hidden padded to 8 x 16 (one N-tile per wave)
constexpr int kBT = 16;    // batch rows per workgroup (one WMMA M-tile)

__device__ __forceinline__ v8f wmma_f16(v16h a, v16h b, v8f c) {
  // D = A x B + C, f32 accumulate  ->  v_wmma_f32_16x16x32_f16
  return __builtin_amdgcn_wmma_f32_16x16x32_f16(
      /*neg_a=*/false, a, /*neg_b=*/false, b,
      /*c_mod=*/(short)0, c, /*reuse_a=*/false, /*reuse_b=*/false);
}

// Branch-free tanh: native v_tanh_f32 if available, else exp2-based
// tanh(x) = 1 - 2/(exp2(2*log2(e)*x) + 1); saturates to +-1 for large |x|.
__device__ __forceinline__ float fast_tanh(float x) {
#if __has_builtin(__builtin_amdgcn_tanhf)
  return __builtin_amdgcn_tanhf(x);
#else
  float e = __builtin_amdgcn_exp2f(x * 2.88539008177793f);
  return 1.0f - 2.0f * __builtin_amdgcn_rcpf(e + 1.0f);
#endif
}

__global__ __launch_bounds__(256) void rnn_fused_kernel(
    const float* __restrict__ x,      // [B, T*I]
    const float* __restrict__ W_ih,   // [H, I]
    const float* __restrict__ W_hh,   // [H, H]
    const float* __restrict__ b_ih,   // [H]
    const float* __restrict__ b_hh,   // [H]
    const float* __restrict__ fc_W,   // [C, H]
    const float* __restrict__ fc_b,   // [C]
    float* __restrict__ out)          // [B, C]
{
  // Double-buffered (ping-pong on t&1) so one barrier/step covers RAW+WAR.
  __shared__ __align__(16) _Float16 xtb[2][kBT * kI];    // step inputs (f16)
  __shared__ __align__(16) _Float16 hbf[2][kBT * kHP];   // hidden state (f16)

  const int tid  = threadIdx.x;
  const int lane = tid & 31;
  const int wv   = tid >> 5;        // wave id 0..7 == N-tile of hidden dim
  const int ln15 = lane & 15;
  const int hi   = lane >> 4;       // which half of the wave (0/1)
  const int bt0  = blockIdx.x * kBT;

  // Column of H (or C in the FC epilogue) owned by this lane in B/C/D layout.
  const int n = wv * 16 + ln15;

  // ---------------------------------------------------------------
  // Build W_hh / W_ih B-matrix register tiles (held for all 64 steps).
  // B (32x16, f16) k order used here (contiguous, not the ISA table order): lane = N column; lanes 0-15 hold K=0..15,
  // lanes 16-31 hold K=16..31; VGPR v holds the K pair (2v, 2v+1).
  // B[k][n] = W^T[k][n] = W[n][k]  (row-major source, zero padded).
  // ---------------------------------------------------------------
  v16h bhh[4];
#pragma unroll
  for (int kk = 0; kk < 4; ++kk) {
    v16h b;
#pragma unroll
    for (int v = 0; v < 8; ++v) {
      int k = kk * 32 + hi * 16 + 2 * v;
      float f0 = (n < kH && k     < kH) ? W_hh[n * kH + k]     : 0.0f;
      float f1 = (n < kH && k + 1 < kH) ? W_hh[n * kH + k + 1] : 0.0f;
      b[2 * v]     = (_Float16)f0;
      b[2 * v + 1] = (_Float16)f1;
    }
    bhh[kk] = b;
  }
  v16h bih[2];
#pragma unroll
  for (int kk = 0; kk < 2; ++kk) {
    v16h b;
#pragma unroll
    for (int v = 0; v < 8; ++v) {
      int k = kk * 32 + hi * 16 + 2 * v;           // always < 64
      float f0 = (n < kH) ? W_ih[n * kI + k]     : 0.0f;
      float f1 = (n < kH) ? W_ih[n * kI + k + 1] : 0.0f;
      b[2 * v]     = (_Float16)f0;
      b[2 * v + 1] = (_Float16)f1;
    }
    bih[kk] = b;
  }
  const float bias = (n < kH) ? (b_ih[n] + b_hh[n]) : 0.0f;

  // ---- zero hidden-state buffer 0 (read at t=0): 2048 halves = 256*16B
  {
    uint4 z{0, 0, 0, 0};
    reinterpret_cast<uint4*>(&hbf[0][0])[tid] = z;
  }

  // ---- A-matrix loader from LDS (16x32 f16 A tile, K offset kbase):
  // lane = M row; lanes 0-15: K 0..15, lanes 16-31: K 16..31
  // (contiguous, matching B).  => two 16B LDS loads per tile.
  auto loadA = [&](const _Float16* buf, int rowStride, int kbase) -> v16h {
    const h8 a0 = *(const h8*)&buf[ln15 * rowStride + kbase + hi * 16];
    const h8 a1 = *(const h8*)&buf[ln15 * rowStride + kbase + 8 + hi * 16];
    v16h a;
#pragma unroll
    for (int j = 0; j < 8; ++j) { a[j] = a0[j]; a[8 + j] = a1[j]; }
    return a;
  };

  // x staging geometry: 16 rows x 64 cols, 4 floats per thread
  const int sm  = tid >> 4;
  const int si4 = (tid & 15) * 4;
  const float* xrow = &x[(size_t)(bt0 + sm) * (kT * kI) + si4];

  // stage t=0 into buffer 0
  {
    f4 v = *(const f4*)&xrow[0];
    h4 o;
#pragma unroll
    for (int j = 0; j < 4; ++j) o[j] = (_Float16)v[j];
    *(h4*)&xtb[0][sm * kI + si4] = o;
  }
  __syncthreads();

  // =========================== recurrence ===========================
  for (int t = 0; t < kT; ++t) {
    const int cur = t & 1, nxt = cur ^ 1;
    const bool more = (t + 1 < kT);

    // issue next step's global x load early; overlaps the WMMA chain
    f4 xv;
    if (more) {
      xv = *(const f4*)&xrow[(t + 1) * kI];
      if (t + 2 < kT)
        __builtin_prefetch(&xrow[(t + 2) * kI], 0, 1);
    }

    v16h ax0 = loadA(xtb[cur], kI,  0);
    v16h ax1 = loadA(xtb[cur], kI,  32);
    v16h ah0 = loadA(hbf[cur], kHP, 0);
    v16h ah1 = loadA(hbf[cur], kHP, 32);
    v16h ah2 = loadA(hbf[cur], kHP, 64);
    v16h ah3 = loadA(hbf[cur], kHP, 96);

    v8f c;
#pragma unroll
    for (int r = 0; r < 8; ++r) c[r] = bias;

    c = wmma_f16(ax0, bih[0], c);
    c = wmma_f16(ax1, bih[1], c);
    c = wmma_f16(ah0, bhh[0], c);
    c = wmma_f16(ah1, bhh[1], c);
    c = wmma_f16(ah2, bhh[2], c);
    c = wmma_f16(ah3, bhh[3], c);

    if (more) {
      h4 o;
#pragma unroll
      for (int j = 0; j < 4; ++j) o[j] = (_Float16)xv[j];
      *(h4*)&xtb[nxt][sm * kI + si4] = o;
    }

    // h_new = tanh(c); C/D layout: VGPR r, lane -> row (r + hi*8), col n
#pragma unroll
    for (int r = 0; r < 8; ++r)
      hbf[nxt][(r + hi * 8) * kHP + n] = (_Float16)fast_tanh(c[r]);

    __syncthreads();   // next step reads buffers [nxt]; WAR safe by ping-pong
  }

  // ====================== epilogue: tanh + FC =======================
  // h_last is in hbf[kT & 1] == hbf[0].  g = tanh(h_last), in place
  // (each thread owns its own 8 halves -> no barrier needed inside).
  {
    h8* p = reinterpret_cast<h8*>(&hbf[0][0]);
    h8 v = p[tid];
    h8 o;
#pragma unroll
    for (int j = 0; j < 8; ++j) o[j] = (_Float16)fast_tanh((float)v[j]);
    p[tid] = o;
  }
  __syncthreads();

  if (wv < 3) {                   // C=40 padded to 48 -> 3 N-tiles
    v16h bfc[4];
#pragma unroll
    for (int kk = 0; kk < 4; ++kk) {
      v16h b;
#pragma unroll
      for (int v = 0; v < 8; ++v) {
        int k = kk * 32 + hi * 16 + 2 * v;
        float f0 = (n < kC && k     < kH) ? fc_W[n * kH + k]     : 0.0f;
        float f1 = (n < kC && k + 1 < kH) ? fc_W[n * kH + k + 1] : 0.0f;
        b[2 * v]     = (_Float16)f0;
        b[2 * v + 1] = (_Float16)f1;
      }
      bfc[kk] = b;
    }
    const float cb = (n < kC) ? fc_b[n] : 0.0f;
    v8f c;
#pragma unroll
    for (int r = 0; r < 8; ++r) c[r] = cb;
#pragma unroll
    for (int kk = 0; kk < 4; ++kk) {
      v16h a = loadA(&hbf[0][0], kHP, kk * 32);
      c = wmma_f16(a, bfc[kk], c);
    }
    if (n < kC) {
#pragma unroll
      for (int r = 0; r < 8; ++r)
        out[(size_t)(bt0 + r + hi * 8) * kC + n] = c[r];
    }
  }
}

extern "C" void kernel_launch(void* const* d_in, const int* in_sizes, int n_in,
                              void* d_out, int out_size, void* d_ws, size_t ws_size,
                              hipStream_t stream) {
  const float* x    = (const float*)d_in[0];
  const float* W_ih = (const float*)d_in[1];
  const float* W_hh = (const float*)d_in[2];
  const float* b_ih = (const float*)d_in[3];
  const float* b_hh = (const float*)d_in[4];
  const float* fc_W = (const float*)d_in[5];
  const float* fc_b = (const float*)d_in[6];
  float* out = (float*)d_out;

  const int B = in_sizes[0] / (kT * kI);   // 16384
  dim3 grid(B / kBT);                      // 1024 workgroups
  dim3 block(256);                         // 8 wave32 waves
  hipLaunchKernelGGL(rnn_fused_kernel, grid, block, 0, stream,
                     x, W_ih, W_hh, b_ih, b_hh, fc_W, fc_b, out);
}
